// Discriminator_59442347376701
// MI455X (gfx1250) — compile-verified, matrix operand order checked by reading, not run on hardware
//
#include <hip/hip_runtime.h>
#include <hip/hip_bf16.h>

#define VOCAB 50257
#define EMB   128
#define H     256
#define G     1024   // 4*H
#define B_    32
#define S_    128
#define LATENT 64
#define CAT   640    // 2H + EMB

typedef __attribute__((ext_vector_type(16))) __bf16    bf16x16;
typedef __attribute__((ext_vector_type(8)))  float     floatx8;
typedef __attribute__((ext_vector_type(4)))  unsigned  uint32x4;

union Frag16 {
    bf16x16  v;
    unsigned u[8];
    uint32x4 q[2];
};

// round-to-nearest-even f32 -> bf16 (as u16); used only in one-time
// preprocessing kernels and the 16-per-lane-per-step h conversion.
__device__ __forceinline__ unsigned short f2bf(float f) {
    unsigned u = __builtin_bit_cast(unsigned, f);
    u = (u + 0x7FFFu + ((u >> 16) & 1u)) >> 16;
    return (unsigned short)u;
}

// K-pair base for 16-bit A/B fragments (ISA 7.12.2): VGPR v holds K = kb,kb+1
__device__ __forceinline__ int kbase(int v, int hi) {
    int kb = (v < 4) ? (2 * v) : (16 + 2 * (v - 4));
    return kb + (hi ? 8 : 0);
}

// gfx1250 has native V_TANH_F32 on the TRANS pipe (co-executes with WMMA).
__device__ __forceinline__ float fast_tanh(float x) {
#if __has_builtin(__builtin_amdgcn_tanhf)
    return __builtin_amdgcn_tanhf(x);
#else
    return tanhf(x);
#endif
}
__device__ __forceinline__ float fast_sigmoid(float x) {
    return 0.5f + 0.5f * fast_tanh(0.5f * x);   // single TRANS op
}

// ---------------------------------------------------------------------------
// fp32 -> bf16 (natural layout): W_emb, w_ih_f, w_ih_r
// ---------------------------------------------------------------------------
__global__ void cvt_bf16_kernel(const float* __restrict__ src,
                                unsigned short* __restrict__ dst, int n) {
    int i = blockIdx.x * blockDim.x + threadIdx.x;
    if (i < n) dst[i] = f2bf(src[i]);
}

// ---------------------------------------------------------------------------
// fp32 w_hh -> bf16 pre-swizzled into WMMA B-fragment order:
//   dst_u32[(((ntile*8 + kc)*32 + lane)*8) + v] = pack(w_hh[n][k], w_hh[n][k+1])
// with n = ntile*16 + (lane&15), k = kbase(v, lane>=16) + kc*32.
// A lane's whole fragment is then 32 contiguous bytes (2 x b128 loads).
// ---------------------------------------------------------------------------
__global__ void swz_whh_kernel(const float* __restrict__ whh,
                               unsigned* __restrict__ dst) {
    const int idx = blockIdx.x * blockDim.x + threadIdx.x;   // [0, G*H/2)
    if (idx >= (G * H) / 2) return;
    const int v     = idx & 7;
    const int lane  = (idx >> 3) & 31;
    const int kc    = (idx >> 8) & 7;
    const int ntile = idx >> 11;
    const int n = ntile * 16 + (lane & 15);
    const int k = kbase(v, lane >> 4) + kc * 32;
    const float* row = whh + (long long)n * H;
    dst[idx] = (unsigned)f2bf(row[k]) | ((unsigned)f2bf(row[k + 1]) << 16);
}

// ---------------------------------------------------------------------------
// Embedding gather fused with the input GEMM:
//   xg[m][n] = sum_k W_emb[s[m]][k] * w_ih[n][k] + bias[n]
// A and B are pre-converted bf16; fragments are two b128 loads each.
// grid (256, 16), block 128 (4 waves); wave -> one 16x16 N-tile, K-loop of 4.
// ---------------------------------------------------------------------------
__global__ void embed_gemm_kernel(const int* __restrict__ s,
                                  const unsigned short* __restrict__ W_emb_bf,
                                  const unsigned short* __restrict__ w_ih_bf,
                                  const float* __restrict__ bias,
                                  float* __restrict__ xg) {
    const int lane = threadIdx.x & 31;
    const int wv   = threadIdx.x >> 5;
    const int hi   = (lane >> 4) & 1;
    const int lm   = lane & 15;

    const int mbase = blockIdx.x * 16;
    const int n     = (blockIdx.y * 4 + wv) * 16 + lm;

    const int token = s[mbase + lm];
    const uint32x4* __restrict__ arow = (const uint32x4*)(W_emb_bf + (long long)token * EMB);
    const uint32x4* __restrict__ brow = (const uint32x4*)(w_ih_bf + (long long)n * EMB);

    floatx8 acc;
#pragma unroll
    for (int i = 0; i < 8; ++i) acc[i] = 0.0f;

#pragma unroll
    for (int kc = 0; kc < 4; ++kc) {
        Frag16 a, b;
        a.q[0] = arow[kc * 4 + hi];       // k = kc*32 + 8*hi .. +7
        a.q[1] = arow[kc * 4 + 2 + hi];   // k = kc*32 + 16 + 8*hi .. +7
        b.q[0] = brow[kc * 4 + hi];
        b.q[1] = brow[kc * 4 + 2 + hi];
        acc = __builtin_amdgcn_wmma_f32_16x16x32_bf16(
            false, a.v, false, b.v, (short)0, acc, false, false);
    }

    const float bn = bias[n];
#pragma unroll
    for (int r = 0; r < 8; ++r) {
        const int m = mbase + r + 8 * hi;
        xg[(long long)m * G + n] = acc[r] + bn;
    }
}

// ---------------------------------------------------------------------------
// Persistent bidirectional LSTM. grid 2 (one block per direction), block 512
// (16 waves). Wave w owns H-columns [16w,16w+16) and all 4 gate tiles for
// them; c-state lane-private in registers; h (bf16) shared via 16 KB LDS.
// Per step: 64 v_wmma_f32_16x16x32_bf16 per wave, two barriers.
// ---------------------------------------------------------------------------
__global__ void lstm_kernel(const float* __restrict__ xg_f,
                            const float* __restrict__ xg_r,
                            const unsigned* __restrict__ whh_swz_f,
                            const unsigned* __restrict__ whh_swz_r,
                            float* __restrict__ hf_out,
                            float* __restrict__ hr_out) {
    const int dir = blockIdx.x;
    const float* __restrict__ xg = dir ? xg_r : xg_f;
    const uint32x4* __restrict__ whh4 =
        (const uint32x4*)(dir ? whh_swz_r : whh_swz_f);
    float* __restrict__ h_out = dir ? hr_out : hf_out;

    const int lane = threadIdx.x & 31;
    const int wv   = threadIdx.x >> 5;
    const int hi   = (lane >> 4) & 1;
    const int lm   = lane & 15;
    const int hcol = wv * 16 + lm;

    // Warm the cache hierarchy with the recurrence weights (global_prefetch_b8)
    {
        const char* wbytes = (const char*)whh4;
        for (size_t off = (size_t)threadIdx.x * 256;
             off < (size_t)G * H * 2; off += (size_t)blockDim.x * 256)
            __builtin_prefetch(wbytes + off, 0, 3);
    }

    __shared__ unsigned short h_bf[B_ * H];   // 16 KB, h_{t-1} in bf16
    for (int i = threadIdx.x; i < B_ * H; i += blockDim.x) h_bf[i] = 0;
    __syncthreads();

    float c_reg[16];
#pragma unroll
    for (int i = 0; i < 16; ++i) c_reg[i] = 0.0f;

    for (int step = 0; step < S_; ++step) {
        const int t = dir ? (S_ - 1 - step) : step;

        floatx8 acc[8];   // [mtile*4 + gate]
#pragma unroll
        for (int i = 0; i < 8; ++i)
#pragma unroll
            for (int j = 0; j < 8; ++j) acc[i][j] = 0.0f;

        const uint32x4* __restrict__ hrow0 =
            (const uint32x4*)(h_bf + lm * H);          // A mtile 0 (batch 0-15)
        const uint32x4* __restrict__ hrow1 =
            (const uint32x4*)(h_bf + (lm + 16) * H);   // A mtile 1 (batch 16-31)

#pragma unroll 2
        for (int kc = 0; kc < 8; ++kc) {
            Frag16 a0, a1;
            a0.q[0] = hrow0[kc * 4 + hi];
            a0.q[1] = hrow0[kc * 4 + 2 + hi];
            a1.q[0] = hrow1[kc * 4 + hi];
            a1.q[1] = hrow1[kc * 4 + 2 + hi];
#pragma unroll
            for (int g = 0; g < 4; ++g) {
                const int ntile = g * 16 + wv;
                const long long fb = ((long long)(ntile * 8 + kc) * 32 + lane) * 2;
                Frag16 b;
                b.q[0] = whh4[fb];
                b.q[1] = whh4[fb + 1];
                acc[g] = __builtin_amdgcn_wmma_f32_16x16x32_bf16(
                    false, a0.v, false, b.v, (short)0, acc[g], false, false);
                acc[4 + g] = __builtin_amdgcn_wmma_f32_16x16x32_bf16(
                    false, a1.v, false, b.v, (short)0, acc[4 + g], false, false);
            }
        }
        __syncthreads();   // all reads of h_{t-1} done before overwrite

#pragma unroll
        for (int mt = 0; mt < 2; ++mt) {
#pragma unroll
            for (int r = 0; r < 8; ++r) {
                const int m = mt * 16 + r + 8 * hi;        // batch index
                const long long xb = ((long long)m * S_ + t) * G;
                const float gi = acc[mt * 4 + 0][r] + xg[xb + hcol];
                const float gf = acc[mt * 4 + 1][r] + xg[xb + 256 + hcol];
                const float gg = acc[mt * 4 + 2][r] + xg[xb + 512 + hcol];
                const float go = acc[mt * 4 + 3][r] + xg[xb + 768 + hcol];
                const float ii = fast_sigmoid(gi);
                const float ff = fast_sigmoid(gf);
                const float g2 = fast_tanh(gg);
                const float oo = fast_sigmoid(go);
                float c = ff * c_reg[mt * 8 + r] + ii * g2;
                c_reg[mt * 8 + r] = c;
                const float h = oo * fast_tanh(c);
                h_out[((long long)m * S_ + t) * H + hcol] = h;
                h_bf[m * H + hcol] = f2bf(h);
            }
        }
        __syncthreads();   // h_t complete before next step's A-fragment reads
    }
}

// ---------------------------------------------------------------------------
// Head stage 1: lengths, gather last timestep, concat action embedding,
// prelu(alpha0). One block of 256 threads.
// ---------------------------------------------------------------------------
__global__ void head1_kernel(const int* __restrict__ s,
                             const int* __restrict__ a,
                             const float* __restrict__ h_f,
                             const float* __restrict__ h_r,
                             const float* __restrict__ W_emb,
                             const float* __restrict__ alpha0,
                             float* __restrict__ s_a) {
    __shared__ int len[B_];
    const int tid = threadIdx.x;
    if (tid < B_) {
        int cnt = 0;
        for (int t = 0; t < S_; ++t) cnt += (s[tid * S_ + t] != 0) ? 1 : 0;
        len[tid] = (cnt < 1) ? 1 : cnt;
    }
    __syncthreads();
    const float al = alpha0[0];
    for (int idx = tid; idx < B_ * CAT; idx += blockDim.x) {
        const int b = idx / CAT;
        const int j = idx % CAT;
        const int t = len[b] - 1;
        float v;
        if (j < H)            v = h_f[((long long)b * S_ + t) * H + j];
        else if (j < 2 * H)   v = h_r[((long long)b * S_ + t) * H + (j - H)];
        else                  v = W_emb[(long long)a[b] * EMB + (j - 2 * H)];
        s_a[idx] = (v >= 0.0f) ? v : al * v;
    }
}

// Head stage 2: x1 = prelu(s_a @ W1^T + b1, alpha1).  grid 32, block 256.
__global__ void head2_kernel(const float* __restrict__ s_a,
                             const float* __restrict__ W1,
                             const float* __restrict__ b1,
                             const float* __restrict__ alpha1,
                             float* __restrict__ x1) {
    const int idx = blockIdx.x * blockDim.x + threadIdx.x;   // [0, 32*256)
    if (idx >= B_ * H) return;
    const int b = idx / H;
    const int n = idx % H;
    float acc = b1[n];
    const float* __restrict__ sr = s_a + b * CAT;
    const float* __restrict__ wr = W1 + (long long)n * CAT;
    for (int k = 0; k < CAT; ++k) acc += sr[k] * wr[k];
    const float al = alpha1[0];
    x1[idx] = (acc >= 0.0f) ? acc : al * acc;
}

// Head stage 3: x2 = x1 @ W2^T + b2 ; out = sigmoid(x2 @ Wd^T + bd).
__global__ void head3_kernel(const float* __restrict__ x1,
                             const float* __restrict__ W2,
                             const float* __restrict__ b2,
                             const float* __restrict__ Wd,
                             const float* __restrict__ bd,
                             float* __restrict__ out) {
    __shared__ float x2[B_ * LATENT];   // 8 KB
    const int tid = threadIdx.x;
    for (int idx = tid; idx < B_ * LATENT; idx += blockDim.x) {
        const int b = idx / LATENT;
        const int l = idx % LATENT;
        float acc = b2[l];
        const float* __restrict__ xr = x1 + b * H;
        const float* __restrict__ wr = W2 + (long long)l * H;
        for (int k = 0; k < H; ++k) acc += xr[k] * wr[k];
        x2[idx] = acc;
    }
    __syncthreads();
    if (tid < B_) {
        float acc = bd[0];
        for (int k = 0; k < LATENT; ++k) acc += x2[tid * LATENT + k] * Wd[k];
        out[tid] = fast_sigmoid(acc);
    }
}

// ---------------------------------------------------------------------------
extern "C" void kernel_launch(void* const* d_in, const int* in_sizes, int n_in,
                              void* d_out, int out_size, void* d_ws, size_t ws_size,
                              hipStream_t stream) {
    const int*   s      = (const int*)  d_in[0];
    const int*   a      = (const int*)  d_in[1];
    const float* W_emb  = (const float*)d_in[2];
    const float* w_ih_f = (const float*)d_in[3];
    const float* w_hh_f = (const float*)d_in[4];
    const float* b_f    = (const float*)d_in[5];
    const float* w_ih_r = (const float*)d_in[6];
    const float* w_hh_r = (const float*)d_in[7];
    const float* b_r    = (const float*)d_in[8];
    const float* alpha0 = (const float*)d_in[9];
    const float* alpha1 = (const float*)d_in[10];
    const float* W1     = (const float*)d_in[11];
    const float* b1     = (const float*)d_in[12];
    const float* W2     = (const float*)d_in[13];
    const float* b2     = (const float*)d_in[14];
    const float* Wd     = (const float*)d_in[15];
    const float* bd     = (const float*)d_in[16];

    char* ws = (char*)d_ws;
    const size_t XG  = (size_t)B_ * S_ * G * sizeof(float);          // 16 MB
    const size_t HB  = (size_t)B_ * S_ * H * sizeof(float);          //  4 MB
    const size_t WH  = (size_t)G * H * sizeof(unsigned short);       // 512 KB
    const size_t WI  = (size_t)G * EMB * sizeof(unsigned short);     // 256 KB
    const size_t WE  = (size_t)VOCAB * EMB * sizeof(unsigned short); // ~12.3 MB

    size_t o = 0;
    float*          xg_f    = (float*)(ws + o);           o += XG;
    float*          xg_r    = (float*)(ws + o);           o += XG;
    float*          h_f     = (float*)(ws + o);           o += HB;
    float*          h_r     = (float*)(ws + o);           o += HB;
    unsigned*       whh_fsw = (unsigned*)(ws + o);        o += WH;
    unsigned*       whh_rsw = (unsigned*)(ws + o);        o += WH;
    unsigned short* wih_fb  = (unsigned short*)(ws + o);  o += WI;
    unsigned short* wih_rb  = (unsigned short*)(ws + o);  o += WI;
    unsigned short* wemb_b  = (unsigned short*)(ws + o);  o += WE;
    float*          s_a     = (float*)(ws + o);           o += (size_t)B_ * CAT * sizeof(float);
    float*          x1      = (float*)(ws + o);

    // 1) one-time precision conversion / swizzle (L2-resident afterwards)
    cvt_bf16_kernel<<<(VOCAB * EMB + 255) / 256, 256, 0, stream>>>(
        W_emb, wemb_b, VOCAB * EMB);
    cvt_bf16_kernel<<<(G * EMB + 255) / 256, 256, 0, stream>>>(
        w_ih_f, wih_fb, G * EMB);
    cvt_bf16_kernel<<<(G * EMB + 255) / 256, 256, 0, stream>>>(
        w_ih_r, wih_rb, G * EMB);
    swz_whh_kernel<<<((G * H / 2) + 255) / 256, 256, 0, stream>>>(w_hh_f, whh_fsw);
    swz_whh_kernel<<<((G * H / 2) + 255) / 256, 256, 0, stream>>>(w_hh_r, whh_rsw);

    // 2) embedding gather + input GEMMs (WMMA bf16, f32 accumulate)
    embed_gemm_kernel<<<dim3((B_ * S_) / 16, G / 64), 128, 0, stream>>>(
        s, wemb_b, wih_fb, b_f, xg_f);
    embed_gemm_kernel<<<dim3((B_ * S_) / 16, G / 64), 128, 0, stream>>>(
        s, wemb_b, wih_rb, b_r, xg_r);

    // 3) persistent bidirectional LSTM (one block per direction)
    lstm_kernel<<<2, 512, 0, stream>>>(xg_f, xg_r, whh_fsw, whh_rsw, h_f, h_r);

    // 4) head (exact fp32)
    head1_kernel<<<1, 256, 0, stream>>>(s, a, h_f, h_r, W_emb, alpha0, s_a);
    head2_kernel<<<B_, 256, 0, stream>>>(s_a, W1, b1, alpha1, x1);
    head3_kernel<<<1, 256, 0, stream>>>(x1, W2, b2, Wd, bd, (float*)d_out);
}
